// MIEsitmator_2800318677559
// MI455X (gfx1250) — matrix operand order repaired and checked by reading, not compiled, not run on hardware
//
#include <hip/hip_runtime.h>

#define B_   64
#define T_   1000
#define U_   200
#define D_   1024
#define H_   512
#define V_   100
#define C_   101      // V+1 classes, blank = 100
#define S_   401      // 2*U+1 extended states
#define NEGF (-1e30f)

typedef __attribute__((ext_vector_type(16))) __bf16 v16bf;
typedef __attribute__((ext_vector_type(8)))  float  v8f;

union BFrag {
  v16bf v;
  __bf16         bf[16];
  unsigned short us[16];
  unsigned int   u32[8];
};

// ---------------------------------------------------------------------------
// Pack W1 [512,1024] fp32 -> bf16 B-fragments: [nt(32)][ks(32)][lane(32)] x 32B
// Fragment layout per ISA 7.12.2: lane -> col n = nt*16+(lane&15),
// K base kb = (lane>>4)*8, K = kb..kb+7 then 16+kb..16+kb+7 as 8 lo/hi pairs.
// ---------------------------------------------------------------------------
__global__ __launch_bounds__(256) void pack_w1_kernel(
    const float* __restrict__ W1, BFrag* __restrict__ W1f)
{
  int tid = blockIdx.x * 256 + threadIdx.x;
  if (tid >= 32 * 32 * 32) return;
  int lane = tid & 31;
  int ks   = (tid >> 5) & 31;
  int nt   = tid >> 10;
  int n    = nt * 16 + (lane & 15);
  int kb   = (lane >> 4) * 8;
  const float* src = W1 + (long)n * D_ + ks * 32 + kb;
  BFrag b;
#pragma unroll
  for (int p = 0; p < 8; ++p) {
    float2 f = *(const float2*)(src + 2 * p + (p >= 4 ? 8 : 0));
    b.bf[2 * p]     = (__bf16)f.x;
    b.bf[2 * p + 1] = (__bf16)f.y;
  }
  W1f[tid] = b;
}

// W2 [101,512] -> bf16 fragments padded to 112 cols: [nt(7)][ks(16)][lane(32)]
__global__ __launch_bounds__(256) void pack_w2_kernel(
    const float* __restrict__ W2, BFrag* __restrict__ W2f)
{
  int tid = blockIdx.x * 256 + threadIdx.x;
  if (tid >= 7 * 16 * 32) return;
  int lane  = tid & 31;
  int ks    = (tid >> 5) & 15;
  int nt    = tid >> 9;
  int n     = nt * 16 + (lane & 15);       // 0..111
  int kb    = (lane >> 4) * 8;
  bool valid = (n < C_);
  const float* src = W2 + (long)(valid ? n : 0) * H_ + ks * 32 + kb;
  BFrag b;
#pragma unroll
  for (int p = 0; p < 8; ++p) {
    float2 f = *(const float2*)(src + 2 * p + (p >= 4 ? 8 : 0));
    b.bf[2 * p]     = valid ? (__bf16)f.x : (__bf16)0.f;
    b.bf[2 * p + 1] = valid ? (__bf16)f.y : (__bf16)0.f;
  }
  W2f[tid] = b;
}

// ---------------------------------------------------------------------------
// Fused: relu(X @ W1^T + b1) @ W2^T + b2 -> log_softmax -> logp
// One workgroup = 32-row M block. 8 waves; GEMM1: each wave owns 2 m-tiles x
// 4 n-tiles (8 WMMA accumulators); GEMM2: waves 0..6 own one 16-col tile.
// ---------------------------------------------------------------------------
__global__ __launch_bounds__(256) void proj_logsoftmax_kernel(
    const float* __restrict__ X,    // [64000, 1024]
    const BFrag* __restrict__ W1f,  // packed bf16 fragments
    const float* __restrict__ b1,   // [512]
    const BFrag* __restrict__ W2f,  // packed bf16 fragments (zero-padded)
    const float* __restrict__ b2,   // [101]
    float* __restrict__ logp)       // [64000, 101]
{
  __shared__ __bf16 htile[32][H_];           // h block, bf16 (32 KB)
  __shared__ float  ltile[32][112];          // logits block (14 KB)

  const int tid    = threadIdx.x;
  const int wave   = tid >> 5;
  const int lane   = tid & 31;
  const int lane15 = lane & 15;
  const int hi     = lane >> 4;
  const int khalf  = hi * 8;       // A-fragment K sub-offset
  const long m0    = (long)blockIdx.x * 32;

  // ---------------- GEMM1 ----------------
  v8f acc[2][4];
  for (int mt = 0; mt < 2; ++mt)
    for (int i = 0; i < 4; ++i)
      for (int r = 0; r < 8; ++r) acc[mt][i][r] = 0.f;

  for (int kk = 0; kk < D_; kk += 32) {
    BFrag a[2];
#pragma unroll
    for (int mt = 0; mt < 2; ++mt) {
      const float* base = X + (m0 + mt * 16 + lane15) * (long)D_ + kk;
      __builtin_prefetch(base + 32, 0, 3);   // WGP-scope prefetch of next K slice
      float4 f0 = *(const float4*)(base + khalf);
      float4 f1 = *(const float4*)(base + khalf + 4);
      float4 f2 = *(const float4*)(base + 16 + khalf);
      float4 f3 = *(const float4*)(base + 16 + khalf + 4);
      a[mt].bf[0]  = (__bf16)f0.x; a[mt].bf[1]  = (__bf16)f0.y;
      a[mt].bf[2]  = (__bf16)f0.z; a[mt].bf[3]  = (__bf16)f0.w;
      a[mt].bf[4]  = (__bf16)f1.x; a[mt].bf[5]  = (__bf16)f1.y;
      a[mt].bf[6]  = (__bf16)f1.z; a[mt].bf[7]  = (__bf16)f1.w;
      a[mt].bf[8]  = (__bf16)f2.x; a[mt].bf[9]  = (__bf16)f2.y;
      a[mt].bf[10] = (__bf16)f2.z; a[mt].bf[11] = (__bf16)f2.w;
      a[mt].bf[12] = (__bf16)f3.x; a[mt].bf[13] = (__bf16)f3.y;
      a[mt].bf[14] = (__bf16)f3.z; a[mt].bf[15] = (__bf16)f3.w;
    }
    const int ks = kk >> 5;
#pragma unroll
    for (int i = 0; i < 4; ++i) {
      const int nt = wave * 4 + i;
      BFrag b = W1f[((long)nt * 32 + ks) * 32 + lane];
      acc[0][i] = __builtin_amdgcn_wmma_f32_16x16x32_bf16(
          false, a[0].v, false, b.v, (short)0, acc[0][i], false, false);
      acc[1][i] = __builtin_amdgcn_wmma_f32_16x16x32_bf16(
          false, a[1].v, false, b.v, (short)0, acc[1][i], false, false);
    }
  }
  // ReLU + bias -> bf16 h block in LDS
#pragma unroll
  for (int i = 0; i < 4; ++i) {
    const int ncol = (wave * 4 + i) * 16 + lane15;
    const float bias = b1[ncol];
#pragma unroll
    for (int mt = 0; mt < 2; ++mt)
#pragma unroll
      for (int r = 0; r < 8; ++r)
        htile[mt * 16 + r + 8 * hi][ncol] = (__bf16)fmaxf(acc[mt][i][r] + bias, 0.f);
  }
  __syncthreads();

  // ---------------- GEMM2 ----------------
  if (wave < 7) {
    const int ncol = wave * 16 + lane15;       // 0..111 (zeros baked for >=101)
    v8f c2[2];
    for (int mt = 0; mt < 2; ++mt)
      for (int r = 0; r < 8; ++r) c2[mt][r] = 0.f;

    for (int kk = 0; kk < H_; kk += 32) {
      BFrag a[2];
#pragma unroll
      for (int mt = 0; mt < 2; ++mt) {
        const __bf16* hrow = &htile[mt * 16 + lane15][kk];
        uint4 c0 = *(const uint4*)(hrow + khalf);       // K = khalf..khalf+7
        uint4 c1 = *(const uint4*)(hrow + 16 + khalf);  // K = 16+khalf..+7
        a[mt].u32[0] = c0.x; a[mt].u32[1] = c0.y;
        a[mt].u32[2] = c0.z; a[mt].u32[3] = c0.w;
        a[mt].u32[4] = c1.x; a[mt].u32[5] = c1.y;
        a[mt].u32[6] = c1.z; a[mt].u32[7] = c1.w;
      }
      BFrag b = W2f[((long)wave * 16 + (kk >> 5)) * 32 + lane];
      c2[0] = __builtin_amdgcn_wmma_f32_16x16x32_bf16(
          false, a[0].v, false, b.v, (short)0, c2[0], false, false);
      c2[1] = __builtin_amdgcn_wmma_f32_16x16x32_bf16(
          false, a[1].v, false, b.v, (short)0, c2[1], false, false);
    }
    if (ncol < C_) {
      const float bias = b2[ncol];
#pragma unroll
      for (int mt = 0; mt < 2; ++mt)
#pragma unroll
        for (int r = 0; r < 8; ++r)
          ltile[mt * 16 + r + 8 * hi][ncol] = c2[mt][r] + bias;
    }
  }
  __syncthreads();

  // ---------------- log_softmax: 8 threads per row, 32 rows ----------------
  {
    const int row = tid >> 3;   // 0..31
    const int sub = tid & 7;
    float mx = -3.4e38f;
    for (int c = sub; c < C_; c += 8) mx = fmaxf(mx, ltile[row][c]);
    for (int off = 4; off >= 1; off >>= 1) mx = fmaxf(mx, __shfl_xor(mx, off, 8));
    float sum = 0.f;
    for (int c = sub; c < C_; c += 8) sum += __expf(ltile[row][c] - mx);
    for (int off = 4; off >= 1; off >>= 1) sum += __shfl_xor(sum, off, 8);
    const float lz = mx + __logf(sum);
    float* orow = logp + (m0 + row) * (long)C_;
    for (int c = sub; c < C_; c += 8) orow[c] = ltile[row][c] - lz;
  }
}

// ---------------------------------------------------------------------------
__global__ void zero_out_kernel(float* out) {
  if (threadIdx.x == 0 && blockIdx.x == 0) out[0] = 0.f;
}

// ---------------------------------------------------------------------------
// CTC alpha recursion: one workgroup per batch element, 13 wave32s cover
// S=401 states; alpha double-buffered in LDS. The next log-prob row is
// prefetched into LDS with global_load_async_to_lds_b32 (ASYNCcnt) by wave 0
// while all waves compute the current step from LDS.
// ---------------------------------------------------------------------------
__global__ __launch_bounds__(416) void ctc_kernel(
    const float* __restrict__ logp,
    const int* __restrict__ tgt,
    const int* __restrict__ dlen,
    const int* __restrict__ tlen,
    float* __restrict__ out)
{
  __shared__ int           ext[S_];
  __shared__ unsigned char skip[S_];
  __shared__ float         alpha[2][S_];
  __shared__ float         lbuf[2][128];    // double-buffered log-prob rows

  const int b     = blockIdx.x;
  const int tid   = threadIdx.x;
  const int wave  = tid >> 5;
  const int lane  = tid & 31;
  const int inlen = dlen[b];
  const int tl    = tlen[b];

  for (int s = tid; s < S_; s += blockDim.x)
    ext[s] = (s & 1) ? tgt[b * U_ + (s >> 1)] : V_;
  __syncthreads();

  for (int s = tid; s < S_; s += blockDim.x) {
    skip[s] = (unsigned char)((s >= 2) && (s & 1) && (ext[s] != ext[s - 2]));
    float a0 = NEGF;
    if (s == 0) a0 = logp[((long)b * T_) * C_ + V_];
    if (s == 1) a0 = logp[((long)b * T_) * C_ + ext[1]];
    alpha[0][s] = a0;
  }

  // Preload row t=1 into lbuf[1] (async-to-LDS, drained before first barrier).
  if (wave == 0) {
    const float* g = logp + ((long)b * T_ + 1) * C_;
    unsigned lds_base = (unsigned)(unsigned long long)(void*)&lbuf[1][0];
#pragma unroll
    for (int j = 0; j < 4; ++j) {
      int cc = lane + j * 32;
      if (cc < C_) {
        unsigned loff = lds_base + cc * 4u;
        unsigned long long ga = (unsigned long long)(g + cc);
        asm volatile("global_load_async_to_lds_b32 %0, %1, off"
                     :: "v"(loff), "v"(ga) : "memory");
      }
    }
    asm volatile("s_wait_asynccnt 0x0" ::: "memory");
  }

  int cur = 0;
  for (int t = 1; t < inlen; ++t) {   // reference freezes at t >= inlen
    __syncthreads();
    // Prefetch row t+1 into the other buffer while computing step t.
    const bool pf = (wave == 0) && (t + 1 < inlen);
    if (pf) {
      const float* g = logp + ((long)b * T_ + t + 1) * C_;
      unsigned lds_base = (unsigned)(unsigned long long)(void*)&lbuf[(t + 1) & 1][0];
#pragma unroll
      for (int j = 0; j < 4; ++j) {
        int cc = lane + j * 32;
        if (cc < C_) {
          unsigned loff = lds_base + cc * 4u;
          unsigned long long ga = (unsigned long long)(g + cc);
          asm volatile("global_load_async_to_lds_b32 %0, %1, off"
                       :: "v"(loff), "v"(ga) : "memory");
        }
      }
    }
    const float* row = &lbuf[t & 1][0];
    for (int s = tid; s < S_; s += blockDim.x) {
      float a0 = alpha[cur][s];
      float a1 = (s >= 1) ? alpha[cur][s - 1] : NEGF;
      float a2 = (s >= 2 && skip[s]) ? alpha[cur][s - 2] : NEGF;
      float m  = fmaxf(a0, fmaxf(a1, a2));
      float l  = m + __logf(__expf(a0 - m) + __expf(a1 - m) + __expf(a2 - m));
      alpha[cur ^ 1][s] = l + row[ext[s]];
    }
    if (pf) asm volatile("s_wait_asynccnt 0x0" ::: "memory");
    cur ^= 1;
  }
  __syncthreads();

  if (tid == 0) {
    int   Sb = 2 * tl + 1;
    float l1 = alpha[cur][Sb - 1];
    float l2 = alpha[cur][Sb - 2];
    float m  = fmaxf(l1, l2);
    float ll = m + __logf(__expf(l1 - m) + __expf(l2 - m));
    atomicAdd(out, (-ll / (float)inlen) * (1.0f / (float)B_));
  }
}

// ---------------------------------------------------------------------------
extern "C" void kernel_launch(void* const* d_in, const int* in_sizes, int n_in,
                              void* d_out, int out_size, void* d_ws, size_t ws_size,
                              hipStream_t stream) {
  (void)in_sizes; (void)n_in; (void)out_size; (void)ws_size;
  const float* X    = (const float*)d_in[0];
  const int*   tgt  = (const int*)  d_in[1];
  const int*   dlen = (const int*)  d_in[2];
  const int*   tlen = (const int*)  d_in[3];
  const float* W1   = (const float*)d_in[4];
  const float* b1   = (const float*)d_in[5];
  const float* W2   = (const float*)d_in[6];
  const float* b2   = (const float*)d_in[7];
  float*       out  = (float*)d_out;

  char* ws = (char*)d_ws;
  BFrag* W1f  = (BFrag*)ws;                            // 1 MB
  BFrag* W2f  = (BFrag*)(ws + (1 << 20));              // 112 KB (pad to 128 KB)
  float* logp = (float*)(ws + (1 << 20) + (1 << 17));  // 25.9 MB

  pack_w1_kernel<<<(32 * 32 * 32 + 255) / 256, 256, 0, stream>>>(W1, W1f);
  pack_w2_kernel<<<(7 * 16 * 32 + 255) / 256, 256, 0, stream>>>(W2, W2f);
  proj_logsoftmax_kernel<<<(B_ * T_) / 32, 256, 0, stream>>>(X, W1f, b1, W2f, b2, logp);
  zero_out_kernel<<<1, 32, 0, stream>>>(out);
  ctc_kernel<<<B_, 416, 0, stream>>>(logp, tgt, dlen, tlen, out);
}
